// GAU_73151882985867
// MI455X (gfx1250) — matrix operand order repaired and checked by reading, not compiled, not run on hardware
//
#include <hip/hip_runtime.h>
#include <hip/hip_bf16.h>

typedef __attribute__((ext_vector_type(16))) _Float16 v16h;
typedef __attribute__((ext_vector_type(8)))  _Float16 v8h;
typedef __attribute__((ext_vector_type(4)))  _Float16 v4h;
typedef __attribute__((ext_vector_type(8)))  float    v8f;
typedef __attribute__((ext_vector_type(4)))  float    v4f;

#ifndef __has_builtin
#define __has_builtin(x) 0
#endif
#if __has_builtin(__builtin_amdgcn_global_load_async_to_lds_b128) && \
    __has_builtin(__builtin_amdgcn_s_wait_asynccnt)
#define GAU_ASYNC_LDS 1
#else
#define GAU_ASYNC_LDS 0
#endif

// ---------------- problem constants ----------------
constexpr int kBatch = 4;
constexpr int kT  = 2048;
constexpr int kD  = 1024;
constexpr int kDF = 2048;
constexpr int kS  = 128;

// ---------------- GEMM tiling ----------------
constexpr int BM = 128, BN = 128, BK = 64;
constexpr int LDS_STRIDE = BK + 8;   // 72 halves = 144 bytes (16B-aligned rows)

enum { MODE_SILU = 0, MODE_QK = 1, MODE_SCORES = 2, MODE_AV = 3, MODE_OUT = 4 };

template <class X, class Y> struct same { static constexpr bool value = false; };
template <class X> struct same<X, X> { static constexpr bool value = true; };

__device__ __forceinline__ v16h cat16(v8h lo, v8h hi) {
  v16h r;
#pragma unroll
  for (int i = 0; i < 8; ++i) { r[i] = lo[i]; r[i + 8] = hi[i]; }
  return r;
}

__device__ __forceinline__ v8f wmma_f16(v16h a, v16h b, v8f c) {
  // D(f32 16x16) = A(f16 16x32) * B(f16 32x16) + C
  return __builtin_amdgcn_wmma_f32_16x16x32_f16(false, a, false, b, (short)0, c,
                                                false, false);
}

#if GAU_ASYNC_LDS
// Builtin prototype (from hipcc diagnostic): param pointee is gcc-vector
// 'int __attribute__((vector_size(16)))' in AS(1) (src) / AS(3) (dst).
// Generic->AS casts go through integers: generic LDS addr low 32 bits ==
// LDS byte offset (aperture rule); global generic addr == AS(1) addr.
typedef int b128_t __attribute__((vector_size(16)));
typedef __attribute__((address_space(1))) b128_t* global_b128_p;
typedef __attribute__((address_space(3))) b128_t* shared_b128_p;

__device__ __forceinline__ void async_copy16(const void* gsrc, void* lds) {
  __builtin_amdgcn_global_load_async_to_lds_b128(
      (global_b128_p)(unsigned long long)(size_t)gsrc,
      (shared_b128_p)(unsigned int)(size_t)lds,
      0, 0);
}
#endif

// ======== tile staging: fetch (global->regs) / commit (regs->LDS f16) ========
// A tile: BM x BK row-major (stride lda); LDS row-major sA[m][k].
template <typename T> struct TileA;

template <> struct TileA<float> {
  v4f r[8];
  __device__ __forceinline__ void fetch(const float* src, int lda, int tid) {
#pragma unroll
    for (int i = 0; i < 8; ++i) {
      int idx = tid + i * 256, row = idx >> 4, c4 = idx & 15;  // 128 x 16 chunks
      r[i] = *(const v4f*)(src + (size_t)row * lda + c4 * 4);
    }
  }
  __device__ __forceinline__ void commit(_Float16* sA, int tid) const {
#pragma unroll
    for (int i = 0; i < 8; ++i) {
      int idx = tid + i * 256, row = idx >> 4, c4 = idx & 15;
      v4h h;
#pragma unroll
      for (int j = 0; j < 4; ++j) h[j] = (_Float16)r[i][j];
      *(v4h*)(&sA[row * LDS_STRIDE + c4 * 4]) = h;
    }
  }
  __device__ __forceinline__ void issue(const float*, int, _Float16*, int) {}
};

template <> struct TileA<_Float16> {   // also used for N x K f16 B tiles (same shape)
  v8h r[4];
  __device__ __forceinline__ void fetch(const _Float16* src, int lda, int tid) {
#pragma unroll
    for (int i = 0; i < 4; ++i) {
      int idx = tid + i * 256, row = idx >> 3, c8 = idx & 7;   // 128 x 8 chunks
      r[i] = *(const v8h*)(src + (size_t)row * lda + c8 * 8);
    }
  }
  __device__ __forceinline__ void commit(_Float16* sA, int tid) const {
#pragma unroll
    for (int i = 0; i < 4; ++i) {
      int idx = tid + i * 256, row = idx >> 3, c8 = idx & 7;
      *(v8h*)(&sA[row * LDS_STRIDE + c8 * 8]) = r[i];
    }
  }
  __device__ __forceinline__ void issue(const _Float16* src, int lda,
                                        _Float16* sA, int tid) {
#if GAU_ASYNC_LDS
#pragma unroll
    for (int i = 0; i < 4; ++i) {
      int idx = tid + i * 256, row = idx >> 3, c8 = idx & 7;
      async_copy16(src + (size_t)row * lda + c8 * 8,
                   &sA[row * LDS_STRIDE + c8 * 8]);
    }
#endif
  }
};

// B tile: BK x BN row-major (stride ldb); stored transposed sB[n][k].
template <typename T> struct TileB;

template <> struct TileB<float> {
  v4f r[8];
  __device__ __forceinline__ void fetch(const float* src, int ldb, int tid) {
#pragma unroll
    for (int i = 0; i < 8; ++i) {
      int idx = tid + i * 256, kr = idx >> 5, c4 = idx & 31;   // 64 x 32 chunks
      r[i] = *(const v4f*)(src + (size_t)kr * ldb + c4 * 4);
    }
  }
  __device__ __forceinline__ void commit(_Float16* sB, int tid) const {
#pragma unroll
    for (int i = 0; i < 8; ++i) {
      int idx = tid + i * 256, kr = idx >> 5, c4 = idx & 31;
#pragma unroll
      for (int j = 0; j < 4; ++j)
        sB[(c4 * 4 + j) * LDS_STRIDE + kr] = (_Float16)r[i][j];
    }
  }
  __device__ __forceinline__ void issue(const float*, int, _Float16*, int) {}
};

template <> struct TileB<_Float16> {
  v8h r[4];
  __device__ __forceinline__ void fetch(const _Float16* src, int ldb, int tid) {
#pragma unroll
    for (int i = 0; i < 4; ++i) {
      int idx = tid + i * 256, kr = idx >> 4, c8 = idx & 15;   // 64 x 16 chunks
      r[i] = *(const v8h*)(src + (size_t)kr * ldb + c8 * 8);
    }
  }
  __device__ __forceinline__ void commit(_Float16* sB, int tid) const {
#pragma unroll
    for (int i = 0; i < 4; ++i) {
      int idx = tid + i * 256, kr = idx >> 4, c8 = idx & 15;
#pragma unroll
      for (int j = 0; j < 8; ++j)
        sB[(c8 * 8 + j) * LDS_STRIDE + kr] = r[i][j];
    }
  }
  __device__ __forceinline__ void issue(const _Float16*, int, _Float16*, int) {}
};

template <typename T, bool NXK> struct BSel { using type = TileB<T>; };
template <> struct BSel<_Float16, true> { using type = TileA<_Float16>; };

// ---------------- fused GEMM (double-buffered LDS pipeline) ----------------
template <int MODE, typename AT, typename BT>
__global__ __launch_bounds__(256, 1) void gau_gemm(
    const AT* __restrict__ A, const BT* __restrict__ Bm,
    int M, int N, int K, int lda, int ldb,
    long long batchA, long long batchB, long long batchO,
    const float* __restrict__ bias,
    const float* __restrict__ gamma, const float* __restrict__ beta,
    const float* __restrict__ uqk,
    const _Float16* __restrict__ umat,
    const int* __restrict__ length,
    void* __restrict__ out0, void* __restrict__ out1) {
  __shared__ alignas(16) _Float16 sA[2][BM * LDS_STRIDE];
  __shared__ alignas(16) _Float16 sB[2][BN * LDS_STRIDE];

  constexpr bool NXK = (MODE == MODE_SCORES);     // B given as N x K row-major
  constexpr bool ASYNC_A = GAU_ASYNC_LDS && same<AT, _Float16>::value;
  constexpr bool ASYNC_B = GAU_ASYNC_LDS && NXK;

  const int tid  = threadIdx.x;
  const int lane = tid & 31;
  const int wid  = tid >> 5;
  const int g    = lane >> 4;        // half-wave group (ISA lane split)
  const int ml   = lane & 15;
  const int m0w  = (wid & 3) * 32;   // 4 waves along M
  const int n0w  = (wid >> 2) * 64;  // 2 waves along N

  const int n0   = blockIdx.x * BN;
  const int row0 = blockIdx.y * BM;
  const int bi   = blockIdx.z;

  const AT* Ab = A  + (size_t)bi * batchA;
  const BT* Bb = Bm + (size_t)bi * batchB;

  auto ptrA = [&](int kb) { return Ab + (size_t)row0 * lda + kb; };
  auto ptrB = [&](int kb) {
    return NXK ? (Bb + (size_t)n0 * ldb + kb) : (Bb + (size_t)kb * ldb + n0);
  };

  v8f zero = {};
  v8f acc[2][4];
#pragma unroll
  for (int mt = 0; mt < 2; ++mt)
#pragma unroll
    for (int nt = 0; nt < 4; ++nt) acc[mt][nt] = zero;

  TileA<AT> ta;
  typename BSel<BT, NXK>::type tb;

  // ---- prologue: stage tile 0 into buffer 0
  if constexpr (ASYNC_A) ta.issue(ptrA(0), lda, &sA[0][0], tid);
  else { ta.fetch(ptrA(0), lda, tid); ta.commit(&sA[0][0], tid); }
  if constexpr (ASYNC_B) tb.issue(ptrB(0), ldb, &sB[0][0], tid);
  else { tb.fetch(ptrB(0), ldb, tid); tb.commit(&sB[0][0], tid); }

  int cur = 0;
  for (int kb = 0; kb < K; kb += BK) {
#if GAU_ASYNC_LDS
    if constexpr (ASYNC_A || ASYNC_B) __builtin_amdgcn_s_wait_asynccnt(0);
#endif
    __syncthreads();
    const bool hasNext = (kb + BK) < K;
    if (hasNext) {
      // async issues write the other buffer; safe after this barrier
      if constexpr (ASYNC_A) ta.issue(ptrA(kb + BK), lda, &sA[cur ^ 1][0], tid);
      else ta.fetch(ptrA(kb + BK), lda, tid);     // overlap global latency w/ WMMA
      if constexpr (ASYNC_B) tb.issue(ptrB(kb + BK), ldb, &sB[cur ^ 1][0], tid);
      else tb.fetch(ptrB(kb + BK), ldb, tid);
    }

    const _Float16* cA = &sA[cur][0];
    const _Float16* cB = &sB[cur][0];
#pragma unroll
    for (int ks = 0; ks < BK / 32; ++ks) {
      v16h af[2], bf[4];
#pragma unroll
      for (int mt = 0; mt < 2; ++mt) {
        // A 16x32 f16 frag: lanes 0-15 K=0..7/16..23, lanes 16-31 K=8..15/24..31
        const _Float16* p = &cA[(m0w + mt * 16 + ml) * LDS_STRIDE + ks * 32 + g * 8];
        af[mt] = cat16(*(const v8h*)p, *(const v8h*)(p + 16));
      }
#pragma unroll
      for (int nt = 0; nt < 4; ++nt) {
        // B 32x16 f16 frag: lane=n, lanes 0-15 K=0..7/16..23, lanes 16-31 K=8..15/24..31
        const _Float16* p = &cB[(n0w + nt * 16 + ml) * LDS_STRIDE + ks * 32 + g * 8];
        bf[nt] = cat16(*(const v8h*)p, *(const v8h*)(p + 16));
      }
#pragma unroll
      for (int mt = 0; mt < 2; ++mt)
#pragma unroll
        for (int nt = 0; nt < 4; ++nt)
          acc[mt][nt] = wmma_f16(af[mt], bf[nt], acc[mt][nt]);
    }

    if (hasNext) {
      if constexpr (!ASYNC_A) ta.commit(&sA[cur ^ 1][0], tid);
      if constexpr (!ASYNC_B) tb.commit(&sB[cur ^ 1][0], tid);
    }
    cur ^= 1;
  }

  // ---- epilogue: C/D layout: VGPR r, lanes 0-15 -> M=r, lanes 16-31 -> M=r+8
#pragma unroll
  for (int mt = 0; mt < 2; ++mt) {
#pragma unroll
    for (int nt = 0; nt < 4; ++nt) {
#pragma unroll
      for (int r = 0; r < 8; ++r) {
        const int grow = row0 + m0w + mt * 16 + g * 8 + r;
        const int gcol = n0 + n0w + nt * 16 + ml;
        float s = acc[mt][nt][r];
        const size_t oidx = (size_t)bi * batchO + (size_t)grow * N + gcol;
        if constexpr (MODE == MODE_SILU) {
          s += bias[gcol];
          float sv = s / (1.0f + __expf(-s));
          ((_Float16*)out0)[oidx] = (_Float16)sv;
        } else if constexpr (MODE == MODE_QK) {
          s += bias[gcol];
          float z  = s / (1.0f + __expf(-s));
          float qv = (z * gamma[gcol] + beta[gcol]) * 0.08838834764831845f + uqk[gcol];
          float kv = z * gamma[kS + gcol] + beta[kS + gcol];
          ((_Float16*)out0)[oidx] = (_Float16)qv;
          ((_Float16*)out1)[oidx] = (_Float16)kv;
        } else if constexpr (MODE == MODE_SCORES) {
          if (gcol >= length[bi]) s = -1e30f;
          ((float*)out0)[oidx] = s;
        } else if constexpr (MODE == MODE_AV) {
          float uu = (float)umat[oidx];
          ((_Float16*)out0)[oidx] = (_Float16)(s * uu);
        } else {  // MODE_OUT
          ((float*)out0)[oidx] = s + bias[gcol];
        }
      }
    }
  }
}

// ---------------- row softmax (T = 2048) ----------------
__global__ __launch_bounds__(256, 1) void softmax_rows(
    const float* __restrict__ scores, _Float16* __restrict__ attn) {
  constexpr int C = kT / 256;  // 8
  const int row = blockIdx.x;
  const float* src = scores + (size_t)row * kT;
  _Float16* dst = attn + (size_t)row * kT;
  const int tid = threadIdx.x;

  float v[C];
  float mx = -3.0e38f;
#pragma unroll
  for (int i = 0; i < C; ++i) { v[i] = src[tid + i * 256]; mx = fmaxf(mx, v[i]); }

  __shared__ float red[256];
  red[tid] = mx;
  __syncthreads();
  for (int s = 128; s > 0; s >>= 1) {
    if (tid < s) red[tid] = fmaxf(red[tid], red[tid + s]);
    __syncthreads();
  }
  mx = red[0];
  __syncthreads();

  float sum = 0.f;
#pragma unroll
  for (int i = 0; i < C; ++i) { v[i] = __expf(v[i] - mx); sum += v[i]; }
  red[tid] = sum;
  __syncthreads();
  for (int s = 128; s > 0; s >>= 1) {
    if (tid < s) red[tid] += red[tid + s];
    __syncthreads();
  }
  const float inv = 1.0f / red[0];
#pragma unroll
  for (int i = 0; i < C; ++i) dst[tid + i * 256] = (_Float16)(v[i] * inv);
}

// ---------------- host launch ----------------
extern "C" void kernel_launch(void* const* d_in, const int* in_sizes, int n_in,
                              void* d_out, int out_size, void* d_ws, size_t ws_size,
                              hipStream_t stream) {
  const float* x     = (const float*)d_in[0];
  const int*   len   = (const int*)d_in[1];
  const float* Wu_w  = (const float*)d_in[2];
  const float* Wu_b  = (const float*)d_in[3];
  const float* Wv_w  = (const float*)d_in[4];
  const float* Wv_b  = (const float*)d_in[5];
  const float* Wqk_w = (const float*)d_in[6];
  const float* Wqk_b = (const float*)d_in[7];
  const float* Wo_w  = (const float*)d_in[8];
  const float* Wo_b  = (const float*)d_in[9];
  const float* gamma = (const float*)d_in[10];
  const float* beta  = (const float*)d_in[11];
  const float* u_qk  = (const float*)d_in[12];
  float* out = (float*)d_out;

  char* ws = (char*)d_ws;
  size_t off = 0;
  _Float16* u_h = (_Float16*)(ws + off); off += (size_t)kBatch * kT * kDF * 2;
  _Float16* v_h = (_Float16*)(ws + off); off += (size_t)kBatch * kT * kDF * 2;
  _Float16* q_h = (_Float16*)(ws + off); off += (size_t)kBatch * kT * kS * 2;
  _Float16* k_h = (_Float16*)(ws + off); off += (size_t)kBatch * kT * kS * 2;
  _Float16* attn_h = (_Float16*)(ws + off); off += (size_t)kBatch * kT * kT * 2;
  float* scores = (float*)(ws + off);       // kBatch*kT*kT*4 bytes
  _Float16* o_h = (_Float16*)scores;        // reuse: scores dead after softmax

  const dim3 blk(256, 1, 1);
  const int MT = kBatch * kT;  // 8192 token rows

  // 1) u = silu(x @ Wu + b)
  gau_gemm<MODE_SILU, float, float><<<dim3(kDF / BN, MT / BM, 1), blk, 0, stream>>>(
      x, Wu_w, MT, kDF, kD, kD, kDF, 0, 0, 0,
      Wu_b, nullptr, nullptr, nullptr, nullptr, nullptr, u_h, nullptr);
  // 2) v = silu(x @ Wv + b)
  gau_gemm<MODE_SILU, float, float><<<dim3(kDF / BN, MT / BM, 1), blk, 0, stream>>>(
      x, Wv_w, MT, kDF, kD, kD, kDF, 0, 0, 0,
      Wv_b, nullptr, nullptr, nullptr, nullptr, nullptr, v_h, nullptr);
  // 3) z = silu(x @ Wqk + b); q = (z*g0+b0)/sqrt(S)+u_qk; k = z*g1+b1
  gau_gemm<MODE_QK, float, float><<<dim3(kS / BN, MT / BM, 1), blk, 0, stream>>>(
      x, Wqk_w, MT, kS, kD, kD, kS, 0, 0, 0,
      Wqk_b, gamma, beta, u_qk, nullptr, nullptr, q_h, k_h);
  // 4) scores[b] = q[b] @ k[b]^T, masked by length
  gau_gemm<MODE_SCORES, _Float16, _Float16><<<dim3(kT / BN, kT / BM, kBatch), blk, 0, stream>>>(
      q_h, k_h, kT, kT, kS, kS, kS,
      (long long)kT * kS, (long long)kT * kS, (long long)kT * kT,
      nullptr, nullptr, nullptr, nullptr, nullptr, len, scores, nullptr);
  // 5) attn = softmax(scores)
  softmax_rows<<<dim3(kBatch * kT, 1, 1), blk, 0, stream>>>(scores, attn_h);
  // 6) o[b] = u[b] * (attn[b] @ v[b])
  gau_gemm<MODE_AV, _Float16, _Float16><<<dim3(kDF / BN, kT / BM, kBatch), blk, 0, stream>>>(
      attn_h, v_h, kT, kDF, kT, kT, kDF,
      (long long)kT * kT, (long long)kT * kDF, (long long)kT * kDF,
      nullptr, nullptr, nullptr, nullptr, u_h, nullptr, o_h, nullptr);
  // 7) out = o @ Wo + b
  gau_gemm<MODE_OUT, _Float16, float><<<dim3(kD / BN, MT / BM, 1), blk, 0, stream>>>(
      o_h, Wo_w, MT, kD, kDF, kDF, kD, 0, 0, 0,
      Wo_b, nullptr, nullptr, nullptr, nullptr, nullptr, out, nullptr);
}
